// AttLstm_25709674234626
// MI455X (gfx1250) — matrix operand order repaired and checked by reading, not compiled, not run on hardware
//
#include <hip/hip_runtime.h>
#include <hip/hip_bf16.h>
#include <math.h>

typedef __bf16 bf16_t;
typedef __attribute__((ext_vector_type(16))) __bf16 v16bf;
typedef __attribute__((ext_vector_type(8)))  float  v8f;

#define B_    256
#define T_    2048
#define D_    128
#define H_    128
#define G_    512      // 4*H
#define MTILE 16       // batch rows per workgroup
#define NT_SCAN 1024   // 32 waves: one 16-wide gate-column tile per wave
#define NT_ATT  256

__device__ __forceinline__ float sigmoid_fast(float x) {
  return __builtin_amdgcn_rcpf(1.0f + __expf(-x));
}
__device__ __forceinline__ float tanh_fast(float x) {
  return 2.0f * __builtin_amdgcn_rcpf(1.0f + __expf(-2.0f * x)) - 1.0f;
}
__device__ __forceinline__ unsigned short f2bf_bits(float f) {
  unsigned u = __float_as_uint(f);
  u += 0x7FFFu + ((u >> 16) & 1u);          // round-to-nearest-even
  return (unsigned short)(u >> 16);
}
__device__ __forceinline__ unsigned f2bf_pack(float lo, float hi) {
  return (unsigned)f2bf_bits(lo) | ((unsigned)f2bf_bits(hi) << 16);
}
__device__ __forceinline__ float bf2f_bits(unsigned short s) {
  return __uint_as_float(((unsigned)s) << 16);
}

// ---------------------------------------------------------------------------
// Scan kernel: 16 workgroups x 1024 threads (32 waves). Each WG owns 16 batch
// rows for the entire T=2048 recurrence (batch-parallel: no inter-WG sync).
// Wave w owns gate columns [16w,16w+16); W_ih/W_hh B-fragments stay in VGPRs.
// Two independent WMMA accumulator chains hide the bf16 WMMA RAW hazard.
// ---------------------------------------------------------------------------
__global__ __launch_bounds__(NT_SCAN) void attlstm_scan(
    const float* __restrict__ x,     // [B,T,D]
    const float* __restrict__ W_ih,  // [4H,D]
    const float* __restrict__ W_hh,  // [4H,H]
    const float* __restrict__ b_ih,  // [4H]
    const float* __restrict__ b_hh,  // [4H]
    unsigned short* __restrict__ hs, // ws: [B,T,H] bf16 bits
    float* __restrict__ hlast)       // ws: [B,H] f32
{
  __shared__ bf16_t x_lds[MTILE * D_];       // 4 KB
  __shared__ bf16_t h_lds[MTILE * H_];       // 4 KB
  __shared__ float  gates_lds[MTILE * G_];   // 32 KB

  const int tid  = threadIdx.x;
  const int w    = tid >> 5;        // wave id == gate-column tile id (0..31)
  const int lane = tid & 31;
  const int m    = lane & 15;
  const int hi   = lane >> 4;
  const int b0   = blockIdx.x * MTILE;

  // B-operand fragments (32x16, contiguous k order, same as A): lane -> column N,
  // element j -> K = kt*32 + hi*16 + j.  gates = A * W^T, so B[k][n] = W[n][k].
  v16bf Bih[4], Bhh[4];
  {
    const int n = w * 16 + m;
#pragma unroll
    for (int kt = 0; kt < 4; ++kt) {
#pragma unroll
      for (int j = 0; j < 16; ++j) {
        const int K = kt * 32 + hi * 16 + j;
        Bih[kt][j] = (bf16_t)W_ih[n * D_ + K];
        Bhh[kt][j] = (bf16_t)W_hh[n * H_ + K];
      }
    }
  }

  // Per-thread elementwise assignment: elements 2*tid, 2*tid+1 of the
  // [16 x 128] h/c tile. Same assignment is used for x staging.
  const int e0 = tid * 2;
  const int er = e0 >> 7;          // row (batch) 0..15
  const int ec = e0 & 127;         // even column
  float bias_r[4][2];
#pragma unroll
  for (int g = 0; g < 4; ++g) {
#pragma unroll
    for (int p = 0; p < 2; ++p)
      bias_r[g][p] = b_ih[g * H_ + ec + p] + b_hh[g * H_ + ec + p];
  }
  float c0 = 0.0f, c1 = 0.0f;

  // zero initial h (dword-packed; e0 covers all even offsets exactly once)
  *(unsigned*)&h_lds[e0] = 0u;

  const float* xptr = x + (((size_t)(b0 + er)) * T_) * D_ + ec;           // +=D_ per t
  unsigned short* hsptr = hs + (((size_t)(b0 + er)) * T_) * H_ + ec;      // +=H_ per t

  for (int t = 0; t < T_; ++t) {
    // ---- stage x_t tile into LDS (packed bf16 pair per thread) ----
    const float2 xv = *(const float2*)xptr;
    *(unsigned*)&x_lds[e0] = f2bf_pack(xv.x, xv.y);
    if (t + 1 < T_) __builtin_prefetch(xptr + D_, 0, 3);   // global_prefetch_b8
    __syncthreads();

    // ---- preload all A fragments, then 2 independent WMMA chains ----
    v16bf ax[4], ah[4];
#pragma unroll
    for (int kt = 0; kt < 4; ++kt) {
#pragma unroll
      for (int j = 0; j < 16; ++j) {
        const int K = kt * 32 + hi * 16 + j;
        ax[kt][j] = x_lds[m * D_ + K];
        ah[kt][j] = h_lds[m * H_ + K];
      }
    }
    v8f acc0 = {}, acc1 = {};
    acc0 = __builtin_amdgcn_wmma_f32_16x16x32_bf16(false, ax[0], false, Bih[0], (short)0, acc0, false, false);
    acc1 = __builtin_amdgcn_wmma_f32_16x16x32_bf16(false, ax[1], false, Bih[1], (short)0, acc1, false, false);
    acc0 = __builtin_amdgcn_wmma_f32_16x16x32_bf16(false, ax[2], false, Bih[2], (short)0, acc0, false, false);
    acc1 = __builtin_amdgcn_wmma_f32_16x16x32_bf16(false, ax[3], false, Bih[3], (short)0, acc1, false, false);
    acc0 = __builtin_amdgcn_wmma_f32_16x16x32_bf16(false, ah[0], false, Bhh[0], (short)0, acc0, false, false);
    acc1 = __builtin_amdgcn_wmma_f32_16x16x32_bf16(false, ah[1], false, Bhh[1], (short)0, acc1, false, false);
    acc0 = __builtin_amdgcn_wmma_f32_16x16x32_bf16(false, ah[2], false, Bhh[2], (short)0, acc0, false, false);
    acc1 = __builtin_amdgcn_wmma_f32_16x16x32_bf16(false, ah[3], false, Bhh[3], (short)0, acc1, false, false);

    // D fragment -> gates LDS (lanes 0-15: M=r8; lanes 16-31: M=r8+8; N=lane&15)
#pragma unroll
    for (int r8 = 0; r8 < 8; ++r8) {
      const int row = r8 + hi * 8;
      const int col = w * 16 + m;
      gates_lds[row * G_ + col] = acc0[r8] + acc1[r8];
    }
    __syncthreads();

    // ---- elementwise LSTM cell update (i,f,g,o), 2 adjacent columns ----
    const float* gr = &gates_lds[er * G_ + ec];
    const float iv0 = sigmoid_fast(gr[0]          + bias_r[0][0]);
    const float iv1 = sigmoid_fast(gr[1]          + bias_r[0][1]);
    const float fv0 = sigmoid_fast(gr[H_]         + bias_r[1][0]);
    const float fv1 = sigmoid_fast(gr[H_ + 1]     + bias_r[1][1]);
    const float gv0 = tanh_fast  (gr[2 * H_]      + bias_r[2][0]);
    const float gv1 = tanh_fast  (gr[2 * H_ + 1]  + bias_r[2][1]);
    const float ov0 = sigmoid_fast(gr[3 * H_]     + bias_r[3][0]);
    const float ov1 = sigmoid_fast(gr[3 * H_ + 1] + bias_r[3][1]);
    c0 = fv0 * c0 + iv0 * gv0;
    c1 = fv1 * c1 + iv1 * gv1;
    const float hv0 = ov0 * tanh_fast(c0);
    const float hv1 = ov1 * tanh_fast(c1);
    const unsigned hpk = f2bf_pack(hv0, hv1);
    *(unsigned*)&h_lds[e0] = hpk;
    *(unsigned*)hsptr = hpk;
    if (t == T_ - 1) {
      hlast[(b0 + er) * H_ + ec]     = hv0;
      hlast[(b0 + er) * H_ + ec + 1] = hv1;
    }
    __syncthreads();

    xptr  += D_;
    hsptr += H_;
  }
}

// ---------------------------------------------------------------------------
// Attention pooling: one workgroup per batch row. Scores staged in LDS,
// softmax (max + sum-exp block reductions), then weighted sum over T.
// ---------------------------------------------------------------------------
__global__ __launch_bounds__(NT_ATT) void attlstm_pool(
    const unsigned short* __restrict__ hs,    // [B,T,H] bf16 bits
    const float* __restrict__ hlast,          // [B,H]
    float* __restrict__ out)                  // [B,H]
{
  __shared__ float sc[T_];        // 8 KB
  __shared__ float hl[H_];
  __shared__ float red[NT_ATT];

  const int b = blockIdx.x, tid = threadIdx.x;
  if (tid < H_) hl[tid] = hlast[b * H_ + tid];
  __syncthreads();

  // scores: dot(h_t, h_last)
  for (int t = tid; t < T_; t += NT_ATT) {
    const unsigned short* row = hs + (((size_t)b) * T_ + t) * H_;
    float acc = 0.0f;
    for (int h = 0; h < H_; h += 2) {
      const unsigned pk = *(const unsigned*)(row + h);   // 2 bf16, 4B aligned
      acc += bf2f_bits((unsigned short)pk)         * hl[h];
      acc += bf2f_bits((unsigned short)(pk >> 16)) * hl[h + 1];
    }
    sc[t] = acc;
  }
  __syncthreads();

  // block max
  float mv = -3.402823466e38f;
  for (int t = tid; t < T_; t += NT_ATT) mv = fmaxf(mv, sc[t]);
  red[tid] = mv; __syncthreads();
  for (int s = NT_ATT / 2; s > 0; s >>= 1) {
    if (tid < s) red[tid] = fmaxf(red[tid], red[tid + s]);
    __syncthreads();
  }
  const float M = red[0];
  __syncthreads();

  // exp + block sum
  float sum = 0.0f;
  for (int t = tid; t < T_; t += NT_ATT) { const float e = __expf(sc[t] - M); sc[t] = e; sum += e; }
  red[tid] = sum; __syncthreads();
  for (int s = NT_ATT / 2; s > 0; s >>= 1) {
    if (tid < s) red[tid] += red[tid + s];
    __syncthreads();
  }
  const float inv = 1.0f / red[0];
  __syncthreads();

  // weighted sum over T (coalesced across h)
  const int h = tid & (H_ - 1), half = tid >> 7;
  float acc = 0.0f;
  for (int t = half * (T_ / 2); t < (half + 1) * (T_ / 2); ++t)
    acc += sc[t] * bf2f_bits(hs[(((size_t)b) * T_ + t) * H_ + h]);
  red[tid] = acc; __syncthreads();
  if (half == 0) out[b * H_ + h] = (red[tid] + red[tid + 128]) * inv;
}

extern "C" void kernel_launch(void* const* d_in, const int* in_sizes, int n_in,
                              void* d_out, int out_size, void* d_ws, size_t ws_size,
                              hipStream_t stream) {
  const float* x    = (const float*)d_in[0];
  const float* W_ih = (const float*)d_in[1];
  const float* W_hh = (const float*)d_in[2];
  const float* b_ih = (const float*)d_in[3];
  const float* b_hh = (const float*)d_in[4];

  unsigned short* hs = (unsigned short*)d_ws;                                   // 128 MB
  float* hlast = (float*)((char*)d_ws + (size_t)B_ * T_ * H_ * sizeof(unsigned short));

  attlstm_scan<<<B_ / MTILE, NT_SCAN, 0, stream>>>(x, W_ih, W_hh, b_ih, b_hh, hs, hlast);
  attlstm_pool<<<B_, NT_ATT, 0, stream>>>(hs, hlast, (float*)d_out);
}
